// Attention_15436112461981
// MI455X (gfx1250) — compile-verified, matrix operand order checked by reading, not run on hardware
//
#include <hip/hip_runtime.h>

typedef __attribute__((ext_vector_type(16))) __bf16 v16bf;
typedef __attribute__((ext_vector_type(8)))  __bf16 v8bf;
typedef __attribute__((ext_vector_type(4)))  __bf16 v4bf;
typedef __attribute__((ext_vector_type(8)))  float  v8f;
typedef __attribute__((ext_vector_type(4)))  float  v4f;
typedef unsigned int u32x4 __attribute__((ext_vector_type(4)));
typedef int          i32x4 __attribute__((ext_vector_type(4)));
typedef int          i32x8 __attribute__((ext_vector_type(8)));

static __device__ __forceinline__ __bf16 f2bf(float f) {
  unsigned u = __builtin_bit_cast(unsigned, f);
  unsigned r = u + 0x7FFFu + ((u >> 16) & 1u);   // round-to-nearest-even
  unsigned short h = (unsigned short)(r >> 16);
  return __builtin_bit_cast(__bf16, h);
}

// LDS byte-offset of a generic pointer into a __shared__ array.
static __device__ __forceinline__ unsigned to_lds_off(const void* p) {
  return (unsigned)(unsigned long long)(__attribute__((address_space(3))) const char*)p;
}

// CDNA5 async copy: 16B per lane, global -> LDS, tracked by ASYNCcnt.
static __device__ __forceinline__ void async_copy_b128(unsigned lds_off, const void* g) {
  asm volatile("global_load_async_to_lds_b128 %0, %1, off"
               :: "v"(lds_off), "v"((unsigned long long)g)
               : "memory");
}
#define WAIT_ASYNC(imm) asm volatile("s_wait_asynccnt " #imm ::: "memory")

// ---------------------------------------------------------------------------
// Tensor Data Mover: 2D bf16 tile (tile_rows x tile_cols elems) from global
// (row stride = row_stride_elems) into LDS, inserting 16B pad per 128B row
// (pad_interval=4 -> 32 DWORDs, pad_amount=3 -> 4 DWORDs) => LDS stride 144B
// which matches TPAD=72 bf16.  D# layout per CDNA5 ISA ch.8.
// ---------------------------------------------------------------------------
static __device__ __forceinline__ void tdm_load_2d(
    unsigned lds_byte_off, const void* gaddr, int tile_rows, int tile_cols,
    int row_stride_elems, int tensor_rows) {
  unsigned long long ga = (unsigned long long)gaddr;
  u32x4 g0;
  g0[0] = 1u;                                              // count=1 (valid)
  g0[1] = lds_byte_off;                                    // lds_addr
  g0[2] = (unsigned)(ga & 0xFFFFFFFFull);                  // global_addr[31:0]
  g0[3] = (unsigned)((ga >> 32) & 0x1FFFFFFull) | (2u << 30); // [56:32] | type=2
  i32x8 g1;
  const unsigned td0 = (unsigned)row_stride_elems;         // tensor_dim0
  const unsigned td1 = (unsigned)tensor_rows;              // tensor_dim1
  g1[0] = (int)((1u << 16)        // data_size = 2 bytes
              | (1u << 20)        // pad_enable
              | (4u << 22)        // pad_interval: 32 DWORDs (128B)
              | (3u << 25));      // pad_amount: 4 DWORDs (16B)
  g1[1] = (int)((td0 & 0xFFFFu) << 16);                    // tensor_dim0[15:0]
  g1[2] = (int)((td0 >> 16) | ((td1 & 0xFFFFu) << 16));    // td0[31:16]|td1[15:0]
  g1[3] = (int)((td1 >> 16) | ((unsigned)tile_cols << 16));// td1[31:16]|tile_dim0
  g1[4] = (int)((unsigned)tile_rows & 0xFFFFu);            // tile_dim1 (dim2=0)
  g1[5] = (int)(unsigned)row_stride_elems;                 // tensor_dim0_stride lo
  g1[6] = 0;
  g1[7] = 0;
  i32x4 gz = {0, 0, 0, 0};
#if defined(__clang_major__) && __clang_major__ >= 23
  i32x8 gz8 = {0, 0, 0, 0, 0, 0, 0, 0};
  __builtin_amdgcn_tensor_load_to_lds(g0, g1, gz, gz, gz8, 0);
#else
  __builtin_amdgcn_tensor_load_to_lds(g0, g1, gz, gz, 0);
#endif
}

// Fragment loader for 16-bit A/B operands of V_WMMA_F32_16X16X32_BF16.
// Per ISA 7.12.2: lanes 0-15 hold (row, K = kk+{0..7, 16..23});
//                 lanes 16-31 hold (row, K = kk+{8..15, 24..31}).
static __device__ __forceinline__ v16bf load_frag(const __bf16* tile, int stride,
                                                  int row0, int kk, int lane) {
  const int half = (lane >> 4) & 1;
  const int r = lane & 15;
  const __bf16* p = tile + (row0 + r) * stride + kk + half * 8;
  v8bf lo = *reinterpret_cast<const v8bf*>(p);        // ds_load_b128
  v8bf hi = *reinterpret_cast<const v8bf*>(p + 16);   // ds_load_b128
  v16bf out;
#pragma unroll
  for (int i = 0; i < 8; ++i) { out[i] = lo[i]; out[i + 8] = hi[i]; }
  return out;
}

// ---------------------------------------------------------------------------
// f32 -> bf16 bulk convert
// ---------------------------------------------------------------------------
__global__ __launch_bounds__(256) void cvt_bf16(const float* __restrict__ in,
                                                __bf16* __restrict__ out, int n4) {
  int i = blockIdx.x * 256 + threadIdx.x;
  if (i < n4) {
    v4f f = reinterpret_cast<const v4f*>(in)[i];
    v4bf b;
#pragma unroll
    for (int q = 0; q < 4; ++q) b[q] = f2bf(f[q]);
    reinterpret_cast<v4bf*>(out)[i] = b;
  }
}

// ---------------------------------------------------------------------------
// GEMM: C[M,N] = A[M,K] * W[N,K]^T, bf16 inputs, TDM-staged, double-buffered.
// Output: bf16 (Cb) or f32+bias (Cf).  M%128==0, N%64==0, K%64==0.
// ---------------------------------------------------------------------------
#define BM 128
#define BN 64
#define BK 64
#define TPAD 72   // padded bf16 row stride in LDS (144B = 9 x 16B chunks)

__global__ __launch_bounds__(256) void gemm_bf16(
    const __bf16* __restrict__ A, const __bf16* __restrict__ W,
    const float* __restrict__ bias, float* __restrict__ Cf,
    __bf16* __restrict__ Cb, int M, int N, int K) {
  __shared__ __bf16 As[2][BM * TPAD];
  __shared__ __bf16 Ws[2][BN * TPAD];

  const int tid  = threadIdx.x;
  const int lane = tid & 31;
  const int wave = tid >> 5;      // 0..7
  const int wm = wave & 3;        // wave row 0..3 (32 rows each)
  const int wn = wave >> 2;       // wave col 0..1 (32 cols each)
  // Scalar (SGPR-provable) wave id so TDM issue guards are SALU branches,
  // not EXEC-mask divergence, in the pipelined k-loop.
  const int swave = (int)(__builtin_amdgcn_readfirstlane(threadIdx.x) >> 5);
  const int blockM = blockIdx.x * BM;
  const int blockN = blockIdx.y * BN;
  const int nk = K / BK;

  // one TDM descriptor per tile: wave 0 moves A, wave 1 moves W
  auto stage = [&](int t) {
    const int kb = t * BK;
    const int buf = t & 1;
    if (swave == 0)
      tdm_load_2d(to_lds_off(&As[buf][0]), A + (size_t)blockM * K + kb,
                  BM, BK, K, M);
    else if (swave == 1)
      tdm_load_2d(to_lds_off(&Ws[buf][0]), W + (size_t)blockN * K + kb,
                  BN, BK, K, N);
  };

  v8f acc[2][2] = {};
  stage(0);

  for (int t = 0; t < nk; ++t) {
    const int cur = t & 1;
    if (t + 1 < nk) {
      stage(t + 1);                            // prefetch next tile
      __builtin_amdgcn_s_wait_tensorcnt(1);    // current tile done
    } else {                                   // (no-op for waves 2..7)
      __builtin_amdgcn_s_wait_tensorcnt(0);
    }
    __syncthreads();             // all tiles landed for every wave

    const __bf16* Ab = As[cur];
    const __bf16* Wb = Ws[cur];
#pragma unroll
    for (int kk = 0; kk < BK; kk += 32) {
      v16bf a0 = load_frag(Ab, TPAD, wm * 32 +  0, kk, lane);
      v16bf a1 = load_frag(Ab, TPAD, wm * 32 + 16, kk, lane);
      v16bf b0 = load_frag(Wb, TPAD, wn * 32 +  0, kk, lane);
      v16bf b1 = load_frag(Wb, TPAD, wn * 32 + 16, kk, lane);
      acc[0][0] = __builtin_amdgcn_wmma_f32_16x16x32_bf16(false, a0, false, b0, (short)0, acc[0][0], false, false);
      acc[0][1] = __builtin_amdgcn_wmma_f32_16x16x32_bf16(false, a0, false, b1, (short)0, acc[0][1], false, false);
      acc[1][0] = __builtin_amdgcn_wmma_f32_16x16x32_bf16(false, a1, false, b0, (short)0, acc[1][0], false, false);
      acc[1][1] = __builtin_amdgcn_wmma_f32_16x16x32_bf16(false, a1, false, b1, (short)0, acc[1][1], false, false);
    }
    __syncthreads();             // done reading before buffer is overwritten
  }

  // epilogue: C/D layout — VGPR v holds (M = v + 8*half), N = lane&15
  const int half = (lane >> 4) & 1;
  const int ln = lane & 15;
#pragma unroll
  for (int i = 0; i < 2; ++i)
#pragma unroll
    for (int j = 0; j < 2; ++j) {
      int col = blockN + wn * 32 + j * 16 + ln;
      float bv = bias ? bias[col] : 0.0f;
#pragma unroll
      for (int v = 0; v < 8; ++v) {
        size_t idx = (size_t)(blockM + wm * 32 + i * 16 + v + half * 8) * N + col;
        if (Cb) Cb[idx] = f2bf(acc[i][j][v]);
        else    Cf[idx] = acc[i][j][v] + bv;
      }
    }
}

// ---------------------------------------------------------------------------
// Flash attention: one block per (b, head, 64-row q tile).  4 waves.
// qkv: [4096][3072] bf16 (q|k|v), rel: [2049][16] f32, o: [4096][1024] bf16
// ---------------------------------------------------------------------------
__global__ __launch_bounds__(128) void attn_kernel(
    const __bf16* __restrict__ qkv, const float* __restrict__ rel,
    __bf16* __restrict__ o) {
  constexpr int N = 2048, D = 64, DIM = 1024, DIM3 = 3072, T = 64;
  __shared__ __bf16 Qs [64 * TPAD];
  __shared__ __bf16 Ks [64 * TPAD];
  __shared__ __bf16 Vts[64 * TPAD];   // V transposed: Vts[d][k]
  __shared__ __bf16 Ps [64 * TPAD];
  __shared__ float  Ss [64 * 68];
  __shared__ float  mS[64], lS[64], aS[64];

  const int tid  = threadIdx.x;
  const int lane = tid & 31;
  const int wave = tid >> 5;                 // 0..3 -> 16 q-rows each
  const int qt = blockIdx.x & 31;
  const int h  = (blockIdx.x >> 5) & 15;
  const int bb = blockIdx.x >> 9;
  const int q0 = qt * 64;
  const size_t rowbase = (size_t)bb * N;

  // async-stage Q tile (unscaled; softmax scale applied in f32 later)
#pragma unroll
  for (int i = 0; i < 4; ++i) {
    int e = tid + i * 128;
    int r = e >> 3, c8 = e & 7;
    async_copy_b128(to_lds_off(&Qs[r * TPAD + c8 * 8]),
                    qkv + (rowbase + q0 + r) * DIM3 + h * D + c8 * 8);
  }
  if (tid < 64) { mS[tid] = -3.0e38f; lS[tid] = 0.0f; }

  v8f oacc[4] = {};

  for (int kt = 0; kt < N / T; ++kt) {
    const int k0 = kt * T;
    __syncthreads();                         // prior reads of Ks/Vts/Ps done
    // async-stage K tile
#pragma unroll
    for (int i = 0; i < 4; ++i) {
      int e = tid + i * 128;
      int r = e >> 3, c8 = e & 7;
      async_copy_b128(to_lds_off(&Ks[r * TPAD + c8 * 8]),
                      qkv + (rowbase + k0 + r) * DIM3 + DIM + h * D + c8 * 8);
    }
    // stage V tile transposed (VGPR path; overlaps with async copies)
#pragma unroll
    for (int i = 0; i < 4; ++i) {
      int e = tid + i * 128;
      int r = e >> 3, c8 = e & 7;
      v8bf vv = *reinterpret_cast<const v8bf*>(
          qkv + (rowbase + k0 + r) * DIM3 + 2 * DIM + h * D + c8 * 8);
#pragma unroll
      for (int q = 0; q < 8; ++q) Vts[(c8 * 8 + q) * TPAD + r] = vv[q];
    }
    WAIT_ASYNC(0x0);
    __syncthreads();

    // S = Q K^T : wave computes 16 q-rows x 64 k-cols
    v8f sacc[4] = {};
#pragma unroll
    for (int kk = 0; kk < 64; kk += 32) {
      v16bf a = load_frag(Qs, TPAD, wave * 16, kk, lane);
#pragma unroll
      for (int j = 0; j < 4; ++j) {
        v16bf b = load_frag(Ks, TPAD, j * 16, kk, lane);
        sacc[j] = __builtin_amdgcn_wmma_f32_16x16x32_bf16(false, a, false, b, (short)0, sacc[j], false, false);
      }
    }
    {
      const int half = (lane >> 4) & 1;
      const int ln = lane & 15;
#pragma unroll
      for (int j = 0; j < 4; ++j)
#pragma unroll
        for (int v = 0; v < 8; ++v)
          Ss[(wave * 16 + v + half * 8) * 68 + j * 16 + ln] = sacc[j][v];
    }
    __syncthreads();

    // online softmax with scale + relative-position bias (thread r owns row r)
    if (tid < 64) {
      const int r = tid;
      const int qg = q0 + r;
      float rowm = -3.0e38f;
      for (int c = 0; c < 64; ++c) {
        int dist = (k0 + c) - qg;
        dist = dist < -1024 ? -1024 : (dist > 1024 ? 1024 : dist);
        float s = Ss[r * 68 + c] * 0.125f + rel[(size_t)(dist + 1024) * 16 + h];
        Ss[r * 68 + c] = s;
        rowm = fmaxf(rowm, s);
      }
      float mold = mS[r];
      float mnew = fmaxf(mold, rowm);
      float alpha = __expf(mold - mnew);
      float sum = 0.0f;
      for (int c = 0; c < 64; ++c) {
        float p = __expf(Ss[r * 68 + c] - mnew);
        Ps[r * TPAD + c] = f2bf(p);
        sum += p;
      }
      lS[r] = lS[r] * alpha + sum;
      mS[r] = mnew;
      aS[r] = alpha;
    }
    __syncthreads();

    // O = alpha*O + P V
    {
      const int half = (lane >> 4) & 1;
#pragma unroll
      for (int j = 0; j < 4; ++j)
#pragma unroll
        for (int v = 0; v < 8; ++v)
          oacc[j][v] *= aS[wave * 16 + v + half * 8];
    }
#pragma unroll
    for (int kk = 0; kk < 64; kk += 32) {
      v16bf a = load_frag(Ps, TPAD, wave * 16, kk, lane);
#pragma unroll
      for (int j = 0; j < 4; ++j) {
        v16bf b = load_frag(Vts, TPAD, j * 16, kk, lane);
        oacc[j] = __builtin_amdgcn_wmma_f32_16x16x32_bf16(false, a, false, b, (short)0, oacc[j], false, false);
      }
    }
  }
  __syncthreads();

  // epilogue: O / l, written bf16 head-interleaved [b, n, h*64+d]
  {
    const int half = (lane >> 4) & 1;
    const int ln = lane & 15;
#pragma unroll
    for (int v = 0; v < 8; ++v) {
      int r = wave * 16 + v + half * 8;
      float rcp = 1.0f / lS[r];
      size_t obase = (rowbase + q0 + r) * DIM + h * D;
#pragma unroll
      for (int j = 0; j < 4; ++j)
        o[obase + j * 16 + ln] = f2bf(oacc[j][v] * rcp);
    }
  }
}

// ---------------------------------------------------------------------------
extern "C" void kernel_launch(void* const* d_in, const int* in_sizes, int n_in,
                              void* d_out, int out_size, void* d_ws, size_t ws_size,
                              hipStream_t stream) {
  const float* x    = (const float*)d_in[0];   // [2,2048,1024]
  const float* Wqkv = (const float*)d_in[1];   // [3072,1024]
  const float* Wout = (const float*)d_in[2];   // [1024,1024]
  const float* bout = (const float*)d_in[3];   // [1024]
  const float* rel  = (const float*)d_in[4];   // [2049,16]
  float* out = (float*)d_out;                  // [2,2048,1024]

  // bf16 workspace layout
  __bf16* xb    = (__bf16*)d_ws;               // 4096*1024
  __bf16* wqkvb = xb    + (size_t)4096 * 1024; // 3072*1024
  __bf16* woutb = wqkvb + (size_t)3072 * 1024; // 1024*1024
  __bf16* qkvb  = woutb + (size_t)1024 * 1024; // 4096*3072
  __bf16* attnb = qkvb  + (size_t)4096 * 3072; // 4096*1024

  cvt_bf16<<<dim3((4096 * 1024 / 4 + 255) / 256), dim3(256), 0, stream>>>(x,    xb,    4096 * 1024 / 4);
  cvt_bf16<<<dim3((3072 * 1024 / 4 + 255) / 256), dim3(256), 0, stream>>>(Wqkv, wqkvb, 3072 * 1024 / 4);
  cvt_bf16<<<dim3((1024 * 1024 / 4 + 255) / 256), dim3(256), 0, stream>>>(Wout, woutb, 1024 * 1024 / 4);

  gemm_bf16<<<dim3(4096 / BM, 3072 / BN), dim3(256), 0, stream>>>(
      xb, wqkvb, nullptr, nullptr, qkvb, 4096, 3072, 1024);
  attn_kernel<<<dim3(1024), dim3(128), 0, stream>>>(qkvb, rel, attnb);
  gemm_bf16<<<dim3(4096 / BM, 1024 / BN), dim3(256), 0, stream>>>(
      attnb, woutb, bout, out, nullptr, 4096, 1024, 1024);
}
